// BahdanauAttention_12412455485497
// MI455X (gfx1250) — compile-verified, matrix operand order checked by reading, not run on hardware
//
#include <hip/hip_runtime.h>
#include <hip/hip_bf16.h>
#include <math.h>

#define ENC2 1024
#define DEC  1024
#define ATT  512
#define BB   32
#define SS   2048

typedef __attribute__((ext_vector_type(16))) __bf16 v16bf;
typedef __attribute__((ext_vector_type(8)))  __bf16 bf16x8;
typedef __attribute__((ext_vector_type(8)))  float  v8f;

__device__ __forceinline__ unsigned short f32_to_bf16_rte(float f) {
  union { float f; unsigned u; } v; v.f = f;
  unsigned r = v.u + 0x7FFFu + ((v.u >> 16) & 1u);  // round to nearest even
  return (unsigned short)(r >> 16);
}

// ---------------------------------------------------------------------------
// Kernel 1: Ws = decoder_hidden @ W_w + W_b    (B x ATT, fp32) -> workspace
// ---------------------------------------------------------------------------
__global__ __launch_bounds__(256)
void ws_gemv_kernel(const float* __restrict__ dh,
                    const float* __restrict__ W_w,
                    const float* __restrict__ W_b,
                    float* __restrict__ Ws) {
  int idx = blockIdx.x * blockDim.x + threadIdx.x;   // 0 .. B*ATT-1
  int b = idx >> 9;          // / ATT
  int a = idx & (ATT - 1);
  const float* h = dh + b * DEC;
  float s = W_b[a];
  #pragma unroll 4
  for (int k = 0; k < DEC; ++k) s += h[k] * W_w[k * ATT + a];
  Ws[idx] = s;
}

// ---------------------------------------------------------------------------
// Kernel 2: transpose + convert U_w (ENC2 x ATT f32) -> U_wT (ATT x ENC2 bf16)
// 1 MB result, L2-resident; makes WMMA B-fragment reads contiguous.
// ---------------------------------------------------------------------------
__global__ __launch_bounds__(256)
void uw_transpose_kernel(const float* __restrict__ U_w,
                         unsigned short* __restrict__ U_wT) {
  int a = blockIdx.x;                                // attention column
  for (int k = threadIdx.x; k < ENC2; k += blockDim.x)
    U_wT[(size_t)a * ENC2 + k] = f32_to_bf16_rte(U_w[(size_t)k * ATT + a]);
}

// ---------------------------------------------------------------------------
// Kernel 3: energy(b, s0..s0+15) = tanh(Ws[b] + enc_tile @ U_w + U_b) . v_w
// One workgroup (8 wave32s) per 16 sequence rows. Encoder tile staged to LDS
// as bf16 once. Each wave computes its 4 N-tiles CONCURRENTLY with 4
// independent v_wmma_f32_16x16x32_bf16 accumulator chains (XDL ILP, 4x A
// fragment reuse), then fuses tanh + v_w reduction.
// ---------------------------------------------------------------------------
__global__ __launch_bounds__(256)
void energy_wmma_kernel(const float* __restrict__ enc,
                        const unsigned short* __restrict__ U_wT,
                        const float* __restrict__ Ws,
                        const float* __restrict__ U_b,
                        const float* __restrict__ v_w,
                        float* __restrict__ energy) {
  __shared__ unsigned short sA[16 * ENC2];   // 32 KB bf16 A-tile
  __shared__ float sE[16];                   // per-row energy accumulators

  const int tid  = threadIdx.x;
  const int lane = tid & 31;
  const int wave = tid >> 5;
  const int b    = blockIdx.y;
  const int s0   = blockIdx.x * 16;

  if (tid < 16) sE[tid] = 0.f;

  // Stage 16 contiguous encoder rows (64 KB f32) into LDS as bf16 (32 KB).
  const float4* src = (const float4*)(enc + ((size_t)b * SS + s0) * ENC2);
  for (int i = tid; i < (16 * ENC2) / 4; i += 256) {
    float4 f = src[i];
    int e = i * 4;
    sA[e + 0] = f32_to_bf16_rte(f.x);
    sA[e + 1] = f32_to_bf16_rte(f.y);
    sA[e + 2] = f32_to_bf16_rte(f.z);
    sA[e + 3] = f32_to_bf16_rte(f.w);
  }
  __syncthreads();

  // ISA 16-bit A/B fragment layout (wave32):
  //   lane < 16 : row/col = lane,    K chunks {k0..k0+7, k0+16..k0+23}
  //   lane >= 16: row/col = lane-16, K chunks {k0+8..k0+15, k0+24..k0+31}
  const int row   = lane & 15;
  const int half  = lane >> 4;
  const int ncolL = lane & 15;

  // This wave's 4 N-tiles: columns nt0*16+ncolL .. (+16,+32,+48)
  const int nbase0 = wave * 64;
  const unsigned short* Brow = U_wT + (size_t)(nbase0 + ncolL) * ENC2;

  v8f acc[4];
  #pragma unroll
  for (int j = 0; j < 4; ++j)
    acc[j] = (v8f){0.f, 0.f, 0.f, 0.f, 0.f, 0.f, 0.f, 0.f};

  #pragma unroll 2
  for (int ks = 0; ks < ENC2 / 32; ++ks) {
    const int kb = ks * 32 + half * 8;
    // A fragment from LDS, shared by all 4 accumulator chains.
    bf16x8 a0 = *(const bf16x8*)&sA[row * ENC2 + kb];
    bf16x8 a1 = *(const bf16x8*)&sA[row * ENC2 + kb + 16];
    v16bf av;
    #pragma unroll
    for (int i = 0; i < 8; ++i) { av[i] = a0[i]; av[i + 8] = a1[i]; }

    // 4 B fragments (L2-resident transposed weights), 4 independent WMMAs.
    #pragma unroll
    for (int j = 0; j < 4; ++j) {
      const unsigned short* Bj = Brow + (size_t)(j * 16) * ENC2;
      bf16x8 b0 = *(const bf16x8*)&Bj[kb];
      bf16x8 b1 = *(const bf16x8*)&Bj[kb + 16];
      v16bf bv;
      #pragma unroll
      for (int i = 0; i < 8; ++i) { bv[i] = b0[i]; bv[i + 8] = b1[i]; }
      acc[j] = __builtin_amdgcn_wmma_f32_16x16x32_bf16(
          /*neg_a=*/false, av, /*neg_b=*/false, bv,
          /*c_mod=*/(short)0, acc[j], /*reuse_a=*/false, /*reuse_b=*/false);
    }
  }

  // Fused epilogue: tanh(acc + Ws + U_b) * v_w, accumulated per lane.
  float epart[8];
  #pragma unroll
  for (int v = 0; v < 8; ++v) epart[v] = 0.f;

  const float* Wsb = Ws + (size_t)b * ATT;
  #pragma unroll
  for (int j = 0; j < 4; ++j) {
    const int ncol = nbase0 + j * 16 + ncolL;
    const float addn = Wsb[ncol] + U_b[ncol];
    const float vwn  = v_w[ncol];
    #pragma unroll
    for (int v = 0; v < 8; ++v) {
      float t = tanhf(acc[j][v] + addn);
      epart[v] += t * vwn;
    }
  }

  // Reduce over the 16 lanes of each half (C/D layout keeps halves separate:
  // lanes 0-15 hold M=v, lanes 16-31 hold M=v+8). xor masks < 16 stay in-half.
  #pragma unroll
  for (int off = 1; off < 16; off <<= 1) {
    #pragma unroll
    for (int v = 0; v < 8; ++v) epart[v] += __shfl_xor(epart[v], off, 32);
  }

  // Merge the 8 waves' partial energies via LDS float atomics (ds_add_f32).
  if (lane == 0) {
    #pragma unroll
    for (int v = 0; v < 8; ++v) atomicAdd(&sE[v], epart[v]);
  } else if (lane == 16) {
    #pragma unroll
    for (int v = 0; v < 8; ++v) atomicAdd(&sE[8 + v], epart[v]);
  }
  __syncthreads();

  if (tid < 16) energy[(size_t)b * SS + s0 + tid] = sE[tid];
}

// ---------------------------------------------------------------------------
// Kernel 4: row softmax over S=2048, one block per batch row.
// ---------------------------------------------------------------------------
__global__ __launch_bounds__(256)
void softmax_kernel(const float* __restrict__ energy, float* __restrict__ out) {
  const int b   = blockIdx.x;
  const int tid = threadIdx.x;
  __shared__ float red[256];
  const float* e = energy + (size_t)b * SS;

  float m = -3.402823466e38f;
  for (int s = tid; s < SS; s += 256) m = fmaxf(m, e[s]);
  red[tid] = m;
  __syncthreads();
  for (int off = 128; off > 0; off >>= 1) {
    if (tid < off) red[tid] = fmaxf(red[tid], red[tid + off]);
    __syncthreads();
  }
  const float M = red[0];
  __syncthreads();

  float sum = 0.f;
  for (int s = tid; s < SS; s += 256) sum += __expf(e[s] - M);
  red[tid] = sum;
  __syncthreads();
  for (int off = 128; off > 0; off >>= 1) {
    if (tid < off) red[tid] += red[tid + off];
    __syncthreads();
  }
  const float inv = 1.f / red[0];

  for (int s = tid; s < SS; s += 256)
    out[(size_t)b * SS + s] = __expf(e[s] - M) * inv;
}

// ---------------------------------------------------------------------------
extern "C" void kernel_launch(void* const* d_in, const int* in_sizes, int n_in,
                              void* d_out, int out_size, void* d_ws, size_t ws_size,
                              hipStream_t stream) {
  (void)in_sizes; (void)n_in; (void)out_size; (void)ws_size;

  const float* dh  = (const float*)d_in[0];   // (B, DEC)
  const float* enc = (const float*)d_in[1];   // (B, S, ENC2)
  const float* W_w = (const float*)d_in[2];   // (DEC, ATT)
  const float* W_b = (const float*)d_in[3];   // (ATT,)
  const float* U_w = (const float*)d_in[4];   // (ENC2, ATT)
  const float* U_b = (const float*)d_in[5];   // (ATT,)
  const float* v_w = (const float*)d_in[6];   // (ATT, 1)
  float* out = (float*)d_out;                 // (B, S)

  // Workspace layout (total ~1.32 MB):
  char* ws = (char*)d_ws;
  float*          Ws     = (float*)ws;                                  // 64 KB
  unsigned short* U_wT   = (unsigned short*)(ws + (size_t)BB * ATT * 4);// 1 MB
  float*          energy = (float*)(ws + (size_t)BB * ATT * 4
                                       + (size_t)ATT * ENC2 * 2);       // 256 KB

  ws_gemv_kernel<<<(BB * ATT) / 256, 256, 0, stream>>>(dh, W_w, W_b, Ws);
  uw_transpose_kernel<<<ATT, 256, 0, stream>>>(U_w, U_wT);

  dim3 grid(SS / 16, BB);
  energy_wmma_kernel<<<grid, 256, 0, stream>>>(enc, U_wT, Ws, U_b, v_w, energy);

  softmax_kernel<<<BB, 256, 0, stream>>>(energy, out);
}
